// Graphormer3D_60945585930265
// MI455X (gfx1250) — matrix operand order repaired and checked by reading, not compiled, not run on hardware
//
#include <hip/hip_runtime.h>
#include <hip/hip_bf16.h>

// ---------------- CDNA5 WMMA plumbing ----------------
typedef __attribute__((ext_vector_type(16))) __bf16 v16bf;
typedef __attribute__((ext_vector_type(8)))  float  v8f;

union Frag {
    v16bf v;
    unsigned short u[16];
    uint4 q[2];
};

__device__ __forceinline__ unsigned short f2b(float x) {
    unsigned int u = __float_as_uint(x);
    unsigned int r = (u + 0x7fffu + ((u >> 16) & 1u)) >> 16;
    return (unsigned short)r;
}
__device__ __forceinline__ float b2f(unsigned short b) {
    return __uint_as_float(((unsigned int)b) << 16);
}
__device__ __forceinline__ float gelu_exact(float x) {
    return 0.5f * x * (1.0f + erff(x * 0.7071067811865476f));
}

#define WMMA_BF16(a, b, c) \
    __builtin_amdgcn_wmma_f32_16x16x32_bf16(false, (a), false, (b), (short)0, (c), false, false)

// Load A fragment (16x32 bf16, row-major, stride ld) per ISA 7.12.2.
__device__ __forceinline__ void load_a_frag(Frag& f, const unsigned short* A,
                                            int row, int k0, int ld, int hf) {
    const unsigned short* p = A + (size_t)row * ld + k0 + hf * 8;
    f.q[0] = *(const uint4*)p;
    f.q[1] = *(const uint4*)(p + 16);
}
// Load B fragment (32x16 bf16) from BT stored [N,K] row-major.
__device__ __forceinline__ void load_bt_frag(Frag& f, const unsigned short* BT,
                                             int col, int k0, int ld, int hf) {
    const unsigned short* p = BT + (size_t)col * ld + k0 + hf * 8;
    f.q[0] = *(const uint4*)p;
    f.q[1] = *(const uint4*)(p + 16);
}

// Async copy 16 bytes global -> LDS (CDNA5, tracked by ASYNCcnt).
__device__ __forceinline__ void async_b128(unsigned lds_addr, const void* gptr) {
    asm volatile("global_load_async_to_lds_b128 %0, %1, off"
                 :: "v"(lds_addr), "v"((unsigned long long)(uintptr_t)gptr)
                 : "memory");
}
__device__ __forceinline__ void wait_async0() {
    asm volatile("s_wait_asynccnt 0x0" ::: "memory");
}

// ---------------- Generic bf16 WMMA GEMM: C[M,N] = A[M,K] @ BT[N,K]^T ----------------
// Double-buffered async-to-LDS staging of 64x64 A tile + 32x64 B tile per K-step.
// mode 0: Cf = v ; 1: Cb = bf16(v) ; 2: Cb = bf16(gelu(v)) ; 3: Cf += v   (v includes bias)
__global__ __launch_bounds__(256)
void gemm_bf16_kernel(const unsigned short* __restrict__ A,
                      const unsigned short* __restrict__ BT,
                      const float* __restrict__ bias,
                      float* __restrict__ Cf, unsigned short* __restrict__ Cb,
                      int M, int N, int K, int mode) {
    // buffer layout (bytes): [A tile 64x64 bf16 = 8192][B tile 32x64 bf16 = 4096] x2
    __shared__ unsigned short smem[12288];  // 24 KB
    const int tid = threadIdx.x, wave = tid >> 5, lane = tid & 31;
    const int hf = lane >> 4, r = lane & 15;
    const int rowBlk = blockIdx.y * 64;
    const int colBlk = blockIdx.x * 32;
    const unsigned lds_base = (unsigned)(uintptr_t)smem;  // flat->LDS uses addr[31:0]

    // stage one 64-wide K panel into LDS buffer `buf` (0/1) with async b128 copies
    auto stage = [&](int k0, int buf) {
        const unsigned baseA = lds_base + (unsigned)buf * 12288u;
        const unsigned baseB = baseA + 8192u;
#pragma unroll
        for (int s = 0; s < 2; ++s) {  // A: 64 rows x 8 chunks of 16B, 2 per thread
            const int c = tid + s * 256;
            const int rw = c >> 3, ch = c & 7;
            async_b128(baseA + (unsigned)(rw * 128 + ch * 16),
                       A + (size_t)(rowBlk + rw) * K + k0 + ch * 8);
        }
        {   // B: 32 cols x 8 chunks of 16B, 1 per thread
            const int rw = tid >> 3, ch = tid & 7;
            async_b128(baseB + (unsigned)(rw * 128 + ch * 16),
                       BT + (size_t)(colBlk + rw) * K + k0 + ch * 8);
        }
    };

    stage(0, 0);
    v8f acc = {};
    const int nsteps = K >> 6;
    for (int s = 0; s < nsteps; ++s) {
        wait_async0();        // my async stores for buffer s&1 have landed
        __syncthreads();      // everyone's have landed; prev buffer fully consumed
        if (s + 1 < nsteps) stage((s + 1) << 6, (s + 1) & 1);
        const unsigned short* bufA = smem + (size_t)(s & 1) * 6144;
        const unsigned short* bufB = bufA + 4096;
#pragma unroll
        for (int t = 0; t < 2; ++t) {  // two k=32 sub-steps from LDS
            Frag a, b;
            const unsigned short* pa = bufA + ((wave & 3) * 16 + r) * 64 + t * 32 + hf * 8;
            a.q[0] = *(const uint4*)pa;
            a.q[1] = *(const uint4*)(pa + 16);
            const unsigned short* pb = bufB + ((wave >> 2) * 16 + r) * 64 + t * 32 + hf * 8;
            b.q[0] = *(const uint4*)pb;
            b.q[1] = *(const uint4*)(pb + 16);
            acc = WMMA_BF16(a.v, b.v, acc);
        }
    }

    const int row0 = rowBlk + (wave & 3) * 16;
    const int n = colBlk + (wave >> 2) * 16 + r;
    const float bv = bias ? bias[n] : 0.0f;
#pragma unroll
    for (int idx = 0; idx < 8; ++idx) {
        const int m = row0 + idx + hf * 8;
        const float v = acc[idx] + bv;
        const size_t off = (size_t)m * N + n;
        if (mode == 0)      Cf[off] = v;
        else if (mode == 1) Cb[off] = f2b(v);
        else if (mode == 2) Cb[off] = f2b(gelu_exact(v));
        else                Cf[off] += v;
    }
}

// ---------------- f32 -> bf16 transpose (math B[K_,N_] -> BT[N_,K_]) ----------------
__global__ void transpose_kernel(const float* __restrict__ W, unsigned short* __restrict__ WT,
                                 int K_, int N_) {
    int idx = blockIdx.x * 256 + threadIdx.x;
    if (idx >= K_ * N_) return;
    int n = idx / K_, k = idx - n * K_;
    WT[idx] = f2b(W[(size_t)k * N_ + n]);
}

// ---------------- node embedding init ----------------
__global__ __launch_bounds__(256)
void emb_kernel(const int* __restrict__ atoms, const int* __restrict__ tags,
                const float* __restrict__ aemb, const float* __restrict__ temb,
                float* __restrict__ x) {
    const int token = blockIdx.x;
    const int a = atoms[token], tg = tags[token];
    for (int d = threadIdx.x; d < 768; d += 256)
        x[(size_t)token * 768 + d] = aemb[(size_t)a * 768 + d] + temb[(size_t)tg * 768 + d];
}

// ---------------- fused GBF + g_sum + attn_bias (WMMA over K=128, H=32) ----------------
__global__ __launch_bounds__(256)
void gbf_bias_kernel(const int* __restrict__ atoms, const float* __restrict__ pos,
                     const float* __restrict__ means, const float* __restrict__ stds,
                     const float* __restrict__ gmul, const float* __restrict__ gadd,
                     const unsigned short* __restrict__ bias_wT,
                     const float* __restrict__ bias_b,
                     float* __restrict__ attn_bias, unsigned short* __restrict__ gsum_bf) {
    extern __shared__ unsigned short gA[];  // 256*128 bf16 = 64 KB
    const int blk = blockIdx.x;
    const int b = blk >> 8, i = blk & 255;
    const int tid = threadIdx.x;
    {   // phase 1: thread j computes g[j, 0..127]
        const int j = tid;
        const int et = atoms[b * 256 + i] * 132 + atoms[b * 256 + j];
        const float mul = gmul[et], add = gadd[et];
        const float dx = pos[(b * 256 + j) * 3 + 0] - pos[(b * 256 + i) * 3 + 0];
        const float dy = pos[(b * 256 + j) * 3 + 1] - pos[(b * 256 + i) * 3 + 1];
        const float dz = pos[(b * 256 + j) * 3 + 2] - pos[(b * 256 + i) * 3 + 2];
        const float xg = mul * sqrtf(dx * dx + dy * dy + dz * dz) + add;
        for (int k = 0; k < 128; ++k) {
            const float sd = fabsf(stds[k]) + 1e-5f;
            const float a = (xg - means[k]) / sd;
            gA[j * 128 + k] = f2b(expf(-0.5f * a * a) / (2.5066270f * sd));
        }
    }
    __syncthreads();
    if (tid < 128) {  // phase 2: g_sum over j for this (b,i)
        float s = 0.0f;
        for (int j = 0; j < 256; ++j) s += b2f(gA[j * 128 + tid]);
        gsum_bf[(size_t)(b * 256 + i) * 128 + tid] = f2b(s);
    }
    __syncthreads();
    // phase 3: attn_bias[b,h,i,:] = g[j,:] @ bias_w via WMMA (M=j 256, N=h 32, K=128)
    const int wave = tid >> 5, lane = tid & 31, hf = lane >> 4, r = lane & 15;
    for (int t = 0; t < 4; ++t) {
        const int tile = wave * 4 + t;
        const int j0 = (tile >> 1) * 16, h0 = (tile & 1) * 16;
        v8f acc = {};
        for (int k0 = 0; k0 < 128; k0 += 32) {
            Frag a, bf;
            const unsigned short* pa = &gA[(j0 + r) * 128 + k0 + hf * 8];
            a.q[0] = *(const uint4*)pa;
            a.q[1] = *(const uint4*)(pa + 16);
            load_bt_frag(bf, bias_wT, h0 + r, k0, 128, hf);
            acc = WMMA_BF16(a.v, bf.v, acc);
        }
        const int h = h0 + r;
        const float bb = bias_b[h];
#pragma unroll
        for (int idx = 0; idx < 8; ++idx) {
            const int j = j0 + idx + hf * 8;
            attn_bias[(((size_t)(b * 32 + h)) * 256 + i) * 256 + j] = acc[idx] + bb;
        }
    }
}

// ---------------- LayerNorm (f32 in, bf16 out) ----------------
__global__ __launch_bounds__(256)
void ln_kernel(const float* __restrict__ x, const float* __restrict__ g,
               const float* __restrict__ bt, unsigned short* __restrict__ out) {
    __shared__ float red[256];
    const int row = blockIdx.x, tid = threadIdx.x;
    const float* xr = x + (size_t)row * 768;
    float s = 0.0f;
    for (int i = tid; i < 768; i += 256) s += xr[i];
    red[tid] = s; __syncthreads();
    for (int o = 128; o > 0; o >>= 1) { if (tid < o) red[tid] += red[tid + o]; __syncthreads(); }
    const float mean = red[0] / 768.0f;
    __syncthreads();
    float v = 0.0f;
    for (int i = tid; i < 768; i += 256) { const float d = xr[i] - mean; v += d * d; }
    red[tid] = v; __syncthreads();
    for (int o = 128; o > 0; o >>= 1) { if (tid < o) red[tid] += red[tid + o]; __syncthreads(); }
    const float rstd = rsqrtf(red[0] / 768.0f + 1e-5f);
    for (int i = tid; i < 768; i += 256)
        out[(size_t)row * 768 + i] = f2b((xr[i] - mean) * rstd * g[i] + bt[i]);
}

// ---------------- attention layer: softmax(QK^T*s + bias) @ V ----------------
__global__ __launch_bounds__(128)
void attn_layer_kernel(const unsigned short* __restrict__ qkv,
                       const float* __restrict__ attn_bias,
                       unsigned short* __restrict__ attn_o) {
    __shared__ float s_s[16 * 256];
    __shared__ unsigned short s_p[16 * 256];
    const int gid = blockIdx.x;
    const int i0 = (gid & 15) * 16;
    const int bh = gid >> 4;
    const int h = bh & 31, b = bh >> 5;
    const int tid = threadIdx.x, wave = tid >> 5, lane = tid & 31;
    const int hf = lane >> 4, r = lane & 15;
    const float scale = 0.20412414523193154f;  // 24^-0.5
    Frag qa;
#pragma unroll
    for (int e = 0; e < 16; ++e) {
        const int d = ((e < 8) ? e : 16 + (e - 8)) + hf * 8;
        qa.u[e] = (d < 24) ? qkv[(size_t)(b * 256 + i0 + r) * 2304 + h * 24 + d] : (unsigned short)0;
    }
    for (int t = 0; t < 4; ++t) {
        const int j0 = (wave * 4 + t) * 16;
        Frag kb;
#pragma unroll
        for (int e = 0; e < 16; ++e) {
            const int d = ((e < 8) ? e : 16 + (e - 8)) + hf * 8;
            kb.u[e] = (d < 24) ? qkv[(size_t)(b * 256 + j0 + r) * 2304 + 768 + h * 24 + d]
                               : (unsigned short)0;
        }
        v8f acc = {};
        acc = WMMA_BF16(qa.v, kb.v, acc);
#pragma unroll
        for (int idx = 0; idx < 8; ++idx) {
            const int row = idx + hf * 8;
            s_s[row * 256 + j0 + r] =
                acc[idx] * scale +
                attn_bias[(((size_t)(b * 32 + h)) * 256 + i0 + row) * 256 + j0 + r];
        }
    }
    __syncthreads();
    {   // softmax: 8 lanes per row, wave32 shfl reductions
        const int row = tid >> 3, sub = tid & 7;
        float mx = -3.4e38f;
        for (int j = sub; j < 256; j += 8) mx = fmaxf(mx, s_s[row * 256 + j]);
        mx = fmaxf(mx, __shfl_xor(mx, 1, 32));
        mx = fmaxf(mx, __shfl_xor(mx, 2, 32));
        mx = fmaxf(mx, __shfl_xor(mx, 4, 32));
        float sum = 0.0f;
        for (int j = sub; j < 256; j += 8) sum += expf(s_s[row * 256 + j] - mx);
        sum += __shfl_xor(sum, 1, 32);
        sum += __shfl_xor(sum, 2, 32);
        sum += __shfl_xor(sum, 4, 32);
        const float inv = 1.0f / sum;
        for (int j = sub; j < 256; j += 8)
            s_p[row * 256 + j] = f2b(expf(s_s[row * 256 + j] - mx) * inv);
    }
    __syncthreads();
    if (wave < 2) {  // P@V : M=16 rows, N = d-tile, K = 256 (8 wmma steps)
        const int d0 = wave * 16;
        const int d = d0 + r;
        v8f acc = {};
        for (int k0 = 0; k0 < 256; k0 += 32) {
            Frag pa, vb;
            const unsigned short* pp = &s_p[r * 256 + k0 + hf * 8];
            pa.q[0] = *(const uint4*)pp;
            pa.q[1] = *(const uint4*)(pp + 16);
#pragma unroll
            for (int e = 0; e < 16; ++e) {
                const int j = k0 + ((e < 8) ? e : 16 + (e - 8)) + hf * 8;
                vb.u[e] = (d < 24) ? qkv[(size_t)(b * 256 + j) * 2304 + 1536 + h * 24 + d]
                                   : (unsigned short)0;
            }
            acc = WMMA_BF16(pa.v, vb.v, acc);
        }
        if (d < 24) {
#pragma unroll
            for (int idx = 0; idx < 8; ++idx) {
                const int row = idx + hf * 8;
                attn_o[(size_t)(b * 256 + i0 + row) * 768 + h * 24 + d] = f2b(acc[idx]);
            }
        }
    }
}

// ---------------- t[b,j,h,c] = sum_d f_w[c,h*24+d] * v_head[b,j,h,d] ----------------
__global__ void tvec_kernel(const unsigned short* __restrict__ vh,
                            const float* __restrict__ f_w, float* __restrict__ tvec) {
    const int idx = blockIdx.x * 256 + threadIdx.x;
    if (idx >= 2048 * 96) return;
    const int token = idx / 96, rem = idx % 96, h = rem / 3, c = rem % 3;
    float s = 0.0f;
    for (int d = 0; d < 24; ++d)
        s += b2f(vh[(size_t)token * 768 + h * 24 + d]) * f_w[c * 768 + h * 24 + d];
    tvec[idx] = s;
}

// ---------------- force head: per (b, i-tile) loop h, WMMA scores + dir contraction ----
__global__ __launch_bounds__(128)
void head_kernel(const unsigned short* __restrict__ qh, const unsigned short* __restrict__ kh,
                 const float* __restrict__ attn_bias, const float* __restrict__ pos,
                 const float* __restrict__ tvec, const float* __restrict__ f_b,
                 float* __restrict__ forces) {
    __shared__ float s_s[16 * 256];
    const int blk = blockIdx.x;
    const int b = blk >> 4, i0 = (blk & 15) * 16;
    const int tid = threadIdx.x, wave = tid >> 5, lane = tid & 31;
    const int hf = lane >> 4, r = lane & 15;
    const int row2 = tid >> 3, sub = tid & 7;
    const float scale = 0.20412414523193154f;
    const float pix = pos[(b * 256 + i0 + row2) * 3 + 0];
    const float piy = pos[(b * 256 + i0 + row2) * 3 + 1];
    const float piz = pos[(b * 256 + i0 + row2) * 3 + 2];
    float acc0 = 0.0f, acc1 = 0.0f, acc2 = 0.0f;
    for (int h = 0; h < 32; ++h) {
        Frag qa;
#pragma unroll
        for (int e = 0; e < 16; ++e) {
            const int d = ((e < 8) ? e : 16 + (e - 8)) + hf * 8;
            qa.u[e] = (d < 24) ? qh[(size_t)(b * 256 + i0 + r) * 768 + h * 24 + d]
                               : (unsigned short)0;
        }
        for (int t = 0; t < 4; ++t) {
            const int j0 = (wave * 4 + t) * 16;
            Frag kb;
#pragma unroll
            for (int e = 0; e < 16; ++e) {
                const int d = ((e < 8) ? e : 16 + (e - 8)) + hf * 8;
                kb.u[e] = (d < 24) ? kh[(size_t)(b * 256 + j0 + r) * 768 + h * 24 + d]
                                   : (unsigned short)0;
            }
            v8f acc = {};
            acc = WMMA_BF16(qa.v, kb.v, acc);
#pragma unroll
            for (int idx = 0; idx < 8; ++idx) {
                const int row = idx + hf * 8;
                s_s[row * 256 + j0 + r] =
                    acc[idx] * scale +
                    attn_bias[(((size_t)(b * 32 + h)) * 256 + i0 + row) * 256 + j0 + r];
            }
        }
        __syncthreads();
        float mx = -3.4e38f;
        for (int j = sub; j < 256; j += 8) mx = fmaxf(mx, s_s[row2 * 256 + j]);
        mx = fmaxf(mx, __shfl_xor(mx, 1, 32));
        mx = fmaxf(mx, __shfl_xor(mx, 2, 32));
        mx = fmaxf(mx, __shfl_xor(mx, 4, 32));
        float sum = 0.0f;
        for (int j = sub; j < 256; j += 8) sum += expf(s_s[row2 * 256 + j] - mx);
        sum += __shfl_xor(sum, 1, 32);
        sum += __shfl_xor(sum, 2, 32);
        sum += __shfl_xor(sum, 4, 32);
        const float inv = 1.0f / sum;
        for (int j = sub; j < 256; j += 8) {
            const float p = expf(s_s[row2 * 256 + j] - mx) * inv;
            const float dx = pos[(b * 256 + j) * 3 + 0] - pix;
            const float dy = pos[(b * 256 + j) * 3 + 1] - piy;
            const float dz = pos[(b * 256 + j) * 3 + 2] - piz;
            const float rin = 1.0f / (sqrtf(dx * dx + dy * dy + dz * dz) + 1e-5f);
            const float* tp = &tvec[((size_t)(b * 256 + j) * 32 + h) * 3];
            acc0 += p * dx * rin * tp[0];
            acc1 += p * dy * rin * tp[1];
            acc2 += p * dz * rin * tp[2];
        }
        __syncthreads();
    }
    acc0 += __shfl_xor(acc0, 1, 32); acc0 += __shfl_xor(acc0, 2, 32); acc0 += __shfl_xor(acc0, 4, 32);
    acc1 += __shfl_xor(acc1, 1, 32); acc1 += __shfl_xor(acc1, 2, 32); acc1 += __shfl_xor(acc1, 4, 32);
    acc2 += __shfl_xor(acc2, 1, 32); acc2 += __shfl_xor(acc2, 2, 32); acc2 += __shfl_xor(acc2, 4, 32);
    if (sub == 0) {
        const size_t o = (size_t)(b * 256 + i0 + row2) * 3;
        forces[o + 0] = acc0 + f_b[0];
        forces[o + 1] = acc1 + f_b[1];
        forces[o + 2] = acc2 + f_b[2];
    }
}

// ---------------- RMSD head finalize ----------------
__global__ __launch_bounds__(256)
void rmsd_kernel(const unsigned short* __restrict__ r1, const float* __restrict__ w2,
                 const float* __restrict__ b2, const float* __restrict__ emb,
                 const int* __restrict__ tags, const unsigned char* __restrict__ mask,
                 float* __restrict__ out) {
    __shared__ float red[256];
    const int b = blockIdx.x, n = threadIdx.x;
    const unsigned short* rr = r1 + (size_t)(b * 256 + n) * 768;
    float dot = 0.0f;
    for (int d = 0; d < 768; ++d) dot += b2f(rr[d]) * w2[d];
    float val = (dot + b2[0]) * emb[tags[b * 256 + n]];
    val = mask[b * 256 + n] ? val : 0.0f;
    red[n] = val; __syncthreads();
    for (int o = 128; o > 0; o >>= 1) { if (n < o) red[n] += red[n + o]; __syncthreads(); }
    if (n == 0) out[b] = red[0];
}

// ============================================================================
extern "C" void kernel_launch(void* const* d_in, const int* in_sizes, int n_in,
                              void* d_out, int out_size, void* d_ws, size_t ws_size,
                              hipStream_t stream) {
    (void)in_sizes; (void)n_in; (void)out_size; (void)ws_size;
    const int* atoms = (const int*)d_in[0];
    const int* tags  = (const int*)d_in[1];
    const float* pos = (const float*)d_in[2];
    const unsigned char* mask = (const unsigned char*)d_in[3];
    const float* p_atom_emb = (const float*)d_in[4];
    const float* p_tag_emb  = (const float*)d_in[5];
    const float* p_gbf_means= (const float*)d_in[6];
    const float* p_gbf_stds = (const float*)d_in[7];
    const float* p_gbf_mul  = (const float*)d_in[8];
    const float* p_gbf_bias = (const float*)d_in[9];
    const float* p_bias_w   = (const float*)d_in[10];
    const float* p_bias_b   = (const float*)d_in[11];
    const float* p_edge_w   = (const float*)d_in[12];
    const float* p_edge_b   = (const float*)d_in[13];
    const float* p_ln1_g = (const float*)d_in[14];
    const float* p_ln1_b = (const float*)d_in[15];
    const float* p_qkv_w = (const float*)d_in[16];
    const float* p_qkv_b = (const float*)d_in[17];
    const float* p_out_w = (const float*)d_in[18];
    const float* p_out_b = (const float*)d_in[19];
    const float* p_ln2_g = (const float*)d_in[20];
    const float* p_ln2_b = (const float*)d_in[21];
    const float* p_fc1_w = (const float*)d_in[22];
    const float* p_fc1_b = (const float*)d_in[23];
    const float* p_fc2_w = (const float*)d_in[24];
    const float* p_fc2_b = (const float*)d_in[25];
    const float* p_fln_g = (const float*)d_in[26];
    const float* p_fln_b = (const float*)d_in[27];
    const float* p_rw1   = (const float*)d_in[28];
    const float* p_rb1   = (const float*)d_in[29];
    const float* p_rw2   = (const float*)d_in[30];
    const float* p_rb2   = (const float*)d_in[31];
    const float* p_remb  = (const float*)d_in[32];
    const float* p_qw = (const float*)d_in[33];
    const float* p_qb = (const float*)d_in[34];
    const float* p_kw = (const float*)d_in[35];
    const float* p_kb = (const float*)d_in[36];
    const float* p_vw = (const float*)d_in[37];
    const float* p_vb = (const float*)d_in[38];
    const float* p_fw = (const float*)d_in[39];
    const float* p_fb = (const float*)d_in[40];
    float* out = (float*)d_out;

    // ---- workspace bump allocator ----
    char* wsp = (char*)d_ws;
    size_t off = 0;
    auto alloc = [&](size_t bytes) -> void* {
        void* p = wsp + off;
        off = (off + bytes + 255) & ~(size_t)255;
        return p;
    };
    float* attn_bias = (float*)alloc((size_t)8 * 32 * 256 * 256 * 4);
    float* x         = (float*)alloc((size_t)2048 * 768 * 4);
    unsigned short* h_bf    = (unsigned short*)alloc((size_t)2048 * 768 * 2);
    unsigned short* qkv_bf  = (unsigned short*)alloc((size_t)2048 * 2304 * 2);
    unsigned short* attn_o  = (unsigned short*)alloc((size_t)2048 * 768 * 2);
    unsigned short* ffn_bf  = (unsigned short*)alloc((size_t)2048 * 3072 * 2);
    unsigned short* gsum_bf = (unsigned short*)alloc((size_t)2048 * 128 * 2);
    unsigned short* xln_bf  = (unsigned short*)alloc((size_t)2048 * 768 * 2);
    unsigned short* r1_bf   = (unsigned short*)alloc((size_t)2048 * 768 * 2);
    unsigned short* qh_bf   = (unsigned short*)alloc((size_t)2048 * 768 * 2);
    unsigned short* kh_bf   = (unsigned short*)alloc((size_t)2048 * 768 * 2);
    unsigned short* vh_bf   = (unsigned short*)alloc((size_t)2048 * 768 * 2);
    float* tvec             = (float*)alloc((size_t)2048 * 32 * 3 * 4);
    unsigned short* edge_wT = (unsigned short*)alloc((size_t)768 * 128 * 2);
    unsigned short* bias_wT = (unsigned short*)alloc((size_t)32 * 128 * 2);
    unsigned short* qkv_wT  = (unsigned short*)alloc((size_t)4 * 2304 * 768 * 2);
    unsigned short* out_wT  = (unsigned short*)alloc((size_t)4 * 768 * 768 * 2);
    unsigned short* fc1_wT  = (unsigned short*)alloc((size_t)4 * 3072 * 768 * 2);
    unsigned short* fc2_wT  = (unsigned short*)alloc((size_t)4 * 768 * 3072 * 2);
    unsigned short* rw1T    = (unsigned short*)alloc((size_t)768 * 768 * 2);
    unsigned short* qwT     = (unsigned short*)alloc((size_t)768 * 768 * 2);
    unsigned short* kwT     = (unsigned short*)alloc((size_t)768 * 768 * 2);
    unsigned short* vwT     = (unsigned short*)alloc((size_t)768 * 768 * 2);

    auto tr = [&](const float* W, unsigned short* WT, int K_, int N_) {
        const int tot = K_ * N_;
        transpose_kernel<<<(tot + 255) / 256, 256, 0, stream>>>(W, WT, K_, N_);
    };
    auto gemm = [&](const unsigned short* A, const unsigned short* BT, const float* bias,
                    float* Cf, unsigned short* Cb, int M, int N, int K, int mode) {
        dim3 grid(N / 32, M / 64);
        gemm_bf16_kernel<<<grid, 256, 0, stream>>>(A, BT, bias, Cf, Cb, M, N, K, mode);
    };

    // ---- weight conversion / transposition (bf16, N-major for WMMA B frags) ----
    tr(p_edge_w, edge_wT, 128, 768);
    tr(p_bias_w, bias_wT, 128, 32);
    for (int l = 0; l < 4; ++l) {
        tr(p_qkv_w + (size_t)l * 768 * 2304, qkv_wT + (size_t)l * 2304 * 768, 768, 2304);
        tr(p_out_w + (size_t)l * 768 * 768,  out_wT + (size_t)l * 768 * 768,  768, 768);
        tr(p_fc1_w + (size_t)l * 768 * 3072, fc1_wT + (size_t)l * 3072 * 768, 768, 3072);
        tr(p_fc2_w + (size_t)l * 3072 * 768, fc2_wT + (size_t)l * 768 * 3072, 3072, 768);
    }
    tr(p_rw1, rw1T, 768, 768);
    tr(p_qw, qwT, 768, 768);
    tr(p_kw, kwT, 768, 768);
    tr(p_vw, vwT, 768, 768);

    // ---- node features + pair bias (g never materialized in HBM) ----
    emb_kernel<<<2048, 256, 0, stream>>>(atoms, tags, p_atom_emb, p_tag_emb, x);
    gbf_bias_kernel<<<2048, 256, 65536, stream>>>(atoms, pos, p_gbf_means, p_gbf_stds,
                                                  p_gbf_mul, p_gbf_bias, bias_wT, p_bias_b,
                                                  attn_bias, gsum_bf);
    gemm(gsum_bf, edge_wT, p_edge_b, x, nullptr, 2048, 768, 128, 3);  // x += gsum@edge_w + b

    // ---- transformer layers ----
    for (int l = 0; l < 4; ++l) {
        ln_kernel<<<2048, 256, 0, stream>>>(x, p_ln1_g + l * 768, p_ln1_b + l * 768, h_bf);
        gemm(h_bf, qkv_wT + (size_t)l * 2304 * 768, p_qkv_b + l * 2304,
             nullptr, qkv_bf, 2048, 2304, 768, 1);
        attn_layer_kernel<<<8 * 32 * 16, 128, 0, stream>>>(qkv_bf, attn_bias, attn_o);
        gemm(attn_o, out_wT + (size_t)l * 768 * 768, p_out_b + l * 768,
             x, nullptr, 2048, 768, 768, 3);
        ln_kernel<<<2048, 256, 0, stream>>>(x, p_ln2_g + l * 768, p_ln2_b + l * 768, h_bf);
        gemm(h_bf, fc1_wT + (size_t)l * 3072 * 768, p_fc1_b + l * 3072,
             nullptr, ffn_bf, 2048, 3072, 768, 2);  // gelu
        gemm(ffn_bf, fc2_wT + (size_t)l * 768 * 3072, p_fc2_b + l * 768,
             x, nullptr, 2048, 768, 3072, 3);
    }
    ln_kernel<<<2048, 256, 0, stream>>>(x, p_fln_g, p_fln_b, xln_bf);

    // ---- RMSD head ----
    gemm(xln_bf, rw1T, p_rb1, nullptr, r1_bf, 2048, 768, 768, 2);  // gelu
    rmsd_kernel<<<8, 256, 0, stream>>>(r1_bf, p_rw2, p_rb2, p_remb, tags, mask, out);

    // ---- force head ----
    gemm(xln_bf, qwT, p_qb, nullptr, qh_bf, 2048, 768, 768, 1);
    gemm(xln_bf, kwT, p_kb, nullptr, kh_bf, 2048, 768, 768, 1);
    gemm(xln_bf, vwT, p_vb, nullptr, vh_bf, 2048, 768, 768, 1);
    tvec_kernel<<<(2048 * 96 + 255) / 256, 256, 0, stream>>>(vh_bf, p_fw, tvec);
    head_kernel<<<8 * 16, 128, 0, stream>>>(qh_bf, kh_bf, attn_bias, pos, tvec, p_fb, out + 8);
}
